// LSTMLosses_25658134626337
// MI455X (gfx1250) — matrix operand order repaired and checked by reading, compiled, not run on hardware
//
#include <hip/hip_runtime.h>

// ---------------------------------------------------------------------------
// Problem constants (B=16, T=512, D=256 from the reference)
// ---------------------------------------------------------------------------
#define BB   16
#define TT   512
#define DD   256
#define MDIM (BB * TT)        // 8192 query rows per direction
#define NDIM (BB * (TT + 1))  // 8208 key rows per direction
#define NT   (NDIM / 16)      // 513 N tiles
#define KST  (DD / 32)        // 8 k-steps of 32

typedef __attribute__((ext_vector_type(16))) __bf16 v16bf;
typedef __attribute__((ext_vector_type(8)))  float  v8f;
typedef __attribute__((ext_vector_type(4)))  int    v4i;

#define GAS __attribute__((address_space(1)))
#define LAS __attribute__((address_space(3)))

union Frag { uint4 q[2]; v16bf v; };

__device__ inline unsigned short f2bf_rne(float x) {
    unsigned u = __float_as_uint(x);
    return (unsigned short)((u + 0x7FFFu + ((u >> 16) & 1u)) >> 16);
}
__device__ inline float bf2f(unsigned short h) {
    return __uint_as_float(((unsigned)h) << 16);
}

// CDNA5 async global->LDS path (ASYNCcnt-tracked), with safe fallback.
#if __has_builtin(__builtin_amdgcn_global_load_async_to_lds_b128)
#define ASYNC_COPY 1
__device__ inline void async_b128(const void* g, void* l) {
    __builtin_amdgcn_global_load_async_to_lds_b128(
        (GAS v4i*)(GAS void*)(g),
        (LAS v4i*)(LAS void*)(l),
        0, 0);
}
#else
#define ASYNC_COPY 0
#endif

__device__ inline void wait_async0() {
#if __has_builtin(__builtin_amdgcn_s_wait_asynccnt)
    __builtin_amdgcn_s_wait_asynccnt(0);
#else
    asm volatile("s_wait_asynccnt 0" ::: "memory");
#endif
}

// ---------------------------------------------------------------------------
// Prep: split hidden rows (h_fw / h_bw) into bf16 hi/lo planes
// hbf layout: [dir][hi|lo][MDIM][DD] ushort
// ---------------------------------------------------------------------------
__global__ __launch_bounds__(256) void prep_h_kernel(
    const float* __restrict__ hidden, unsigned short* __restrict__ hbf) {
    size_t idx = (size_t)blockIdx.x * 256 + threadIdx.x; // over 2*MDIM*DD
    if (idx >= (size_t)2 * MDIM * DD) return;
    int d        = (int)(idx & (DD - 1));
    size_t row   = (idx >> 8) & (MDIM - 1);
    int dir      = (int)(idx >> (8 + 13));
    float x      = hidden[row * (2 * DD) + dir * DD + d];
    unsigned short h = f2bf_rne(x);
    unsigned short l = f2bf_rne(x - bf2f(h));
    hbf[((size_t)(dir * 2 + 0) * MDIM + row) * DD + d] = h;
    hbf[((size_t)(dir * 2 + 1) * MDIM + row) * DD + d] = l;
}

// ---------------------------------------------------------------------------
// Prep: build X_fw / X_bw (with the inserted zero rows) in bf16 hi/lo
// xbf layout: [dir][hi|lo][NDIM][DD] ushort
// ---------------------------------------------------------------------------
__global__ __launch_bounds__(256) void prep_x_kernel(
    const float* __restrict__ feats, unsigned short* __restrict__ xbf) {
    size_t idx = (size_t)blockIdx.x * 256 + threadIdx.x; // over 2*NDIM*DD
    if (idx >= (size_t)2 * NDIM * DD) return;
    int d     = (int)(idx & (DD - 1));
    size_t np = idx >> 8;               // 0 .. 2*NDIM-1
    int dir   = (int)(np / NDIM);
    int n     = (int)(np % NDIM);
    int b     = n / (TT + 1);
    int i     = n % (TT + 1);
    float x;
    if (dir == 0) x = (i <  TT) ? feats[((size_t)b * TT + i)       * DD + d] : 0.f;
    else          x = (i == 0 ) ? 0.f : feats[((size_t)b * TT + (i - 1)) * DD + d];
    unsigned short h = f2bf_rne(x);
    unsigned short l = f2bf_rne(x - bf2f(h));
    xbf[((size_t)(dir * 2 + 0) * NDIM + n) * DD + d] = h;
    xbf[((size_t)(dir * 2 + 1) * NDIM + n) * DD + d] = l;
}

// ---------------------------------------------------------------------------
// Fused GEMM + online logsumexp.
// Grid: (MDIM/128, 2).  Block: 256 threads = 8 waves, 16 M-rows per wave.
// A (h rows) held in registers (hi+lo) for the whole N sweep; B (X rows)
// double-buffered through LDS (async direct-to-LDS when available) and
// shared by all 8 waves.  Three WMMA accumulator chains break the RAW
// dependency so the XDL pipe can issue back-to-back.
// ---------------------------------------------------------------------------
#define ROWB 1040                 // 512 hi + 512 lo + 16 pad (bank-conflict-free)
#define BUFB (16 * ROWB)

__global__ __launch_bounds__(256) void gemm_lse_kernel(
    const unsigned short* __restrict__ hbf,
    const unsigned short* __restrict__ xbf,
    float* __restrict__ lse) {
    const int dir = blockIdx.y;
    const unsigned short* Ahi = hbf + (size_t)(dir * 2 + 0) * MDIM * DD;
    const unsigned short* Alo = hbf + (size_t)(dir * 2 + 1) * MDIM * DD;
    const unsigned short* Bhi = xbf + (size_t)(dir * 2 + 0) * NDIM * DD;
    const unsigned short* Blo = xbf + (size_t)(dir * 2 + 1) * NDIM * DD;

    __shared__ __align__(16) unsigned char smem[2 * BUFB];

    const int tid  = threadIdx.x;
    const int wave = tid >> 5;
    const int lane = tid & 31;
    const int l16  = lane & 15;
    const int half = lane >> 4;
    const int mbase = blockIdx.x * 128 + wave * 16;

    // ---- A fragments (documented CDNA5 16-bit A layout):
    // lane<16 : K {k..k+7, k+16..k+23}; lane>=16 : K {k+8..k+15, k+24..k+31}
    Frag ahi[KST], alo[KST];
    {
        const unsigned char* pa =
            (const unsigned char*)(Ahi + (size_t)(mbase + l16) * DD) + half * 16;
        const unsigned char* pb =
            (const unsigned char*)(Alo + (size_t)(mbase + l16) * DD) + half * 16;
#pragma unroll
        for (int ks = 0; ks < KST; ++ks) {
            ahi[ks].q[0] = *(const uint4*)(pa + ks * 64);
            ahi[ks].q[1] = *(const uint4*)(pa + ks * 64 + 32);
            alo[ks].q[0] = *(const uint4*)(pb + ks * 64);
            alo[ks].q[1] = *(const uint4*)(pb + ks * 64 + 32);
        }
    }

#if ASYNC_COPY
    auto load_tile = [&](int nt, int buf) {
        const char* ghi = (const char*)Bhi + (size_t)nt * 16 * 512;
        const char* glo = (const char*)Blo + (size_t)nt * 16 * 512;
#pragma unroll
        for (int it = 0; it < 2; ++it) {
            int c   = tid + it * 256;   // 512 16-byte chunks
            int row = c >> 5;
            int cin = c & 31;
            async_b128(ghi + row * 512 + cin * 16,
                       smem + buf * BUFB + row * ROWB + cin * 16);
            async_b128(glo + row * 512 + cin * 16,
                       smem + buf * BUFB + row * ROWB + 512 + cin * 16);
        }
    };
#else
    auto load_tile = [&](int nt, int buf) {
        const unsigned char* ghi = (const unsigned char*)Bhi + (size_t)nt * 16 * 512;
        const unsigned char* glo = (const unsigned char*)Blo + (size_t)nt * 16 * 512;
#pragma unroll
        for (int it = 0; it < 2; ++it) {
            int c   = tid + it * 256;   // 512 16-byte chunks
            int row = c >> 5;
            int cin = c & 31;
            *(uint4*)(smem + buf * BUFB + row * ROWB + cin * 16) =
                *(const uint4*)(ghi + row * 512 + cin * 16);
            *(uint4*)(smem + buf * BUFB + row * ROWB + 512 + cin * 16) =
                *(const uint4*)(glo + row * 512 + cin * 16);
        }
    };
#endif

    float mrun[8], srun[8];
#pragma unroll
    for (int r = 0; r < 8; ++r) { mrun[r] = -__builtin_inff(); srun[r] = 0.f; }

    load_tile(0, 0);
#if ASYNC_COPY
    wait_async0();
#endif
    __syncthreads();

    for (int nt = 0; nt < NT; ++nt) {
        const int buf = nt & 1;
        if (nt + 1 < NT) load_tile(nt + 1, buf ^ 1);

        // B fragment: lane<16 -> column n, K {k..k+7, k+16..k+23} ; lane>=16 -> K {k+8..k+15, k+24..k+31}
        const unsigned char* bbase = smem + buf * BUFB + l16 * ROWB + half * 16;
        v8f c0 = {}, c1 = {}, c2 = {};
#pragma unroll
        for (int ks = 0; ks < KST; ++ks) {
            Frag bh, bl;
            bh.q[0] = *(const uint4*)(bbase + ks * 64);
            bh.q[1] = *(const uint4*)(bbase + ks * 64 + 32);
            bl.q[0] = *(const uint4*)(bbase + 512 + ks * 64);
            bl.q[1] = *(const uint4*)(bbase + 512 + ks * 64 + 32);
            c0 = __builtin_amdgcn_wmma_f32_16x16x32_bf16(false, ahi[ks].v, false, bh.v,
                                                         (short)0, c0, false, false);
            c1 = __builtin_amdgcn_wmma_f32_16x16x32_bf16(false, ahi[ks].v, false, bl.v,
                                                         (short)0, c1, false, false);
            c2 = __builtin_amdgcn_wmma_f32_16x16x32_bf16(false, alo[ks].v, false, bh.v,
                                                         (short)0, c2, false, false);
        }
        // online logsumexp fold of this 16x16 logits tile (co-executes with WMMA)
#pragma unroll
        for (int r = 0; r < 8; ++r) {
            float v  = c0[r] + c1[r] + c2[r];
            float mo = mrun[r];
            float mn = fmaxf(mo, v);
            srun[r]  = srun[r] * __expf(mo - mn) + __expf(v - mn);
            mrun[r]  = mn;
        }
#if ASYNC_COPY
        wait_async0();
#endif
        __syncthreads();
    }

    // ---- cross-lane reduction: C VGPR r holds row (r + 8*half), col l16
    float* red = (float*)smem + wave * (16 * 16 * 2);
#pragma unroll
    for (int r = 0; r < 8; ++r) {
        int rl = r + half * 8;
        red[(rl * 16 + l16) * 2 + 0] = mrun[r];
        red[(rl * 16 + l16) * 2 + 1] = srun[r];
    }
    __syncthreads();
    if (lane < 16) {
        const float* rr = red + lane * 32;
        float Mv = -__builtin_inff(), Sv = 0.f;
#pragma unroll
        for (int j = 0; j < 16; ++j) {
            float mj = rr[j * 2], sj = rr[j * 2 + 1];
            float mn = fmaxf(Mv, mj);
            Sv = Sv * __expf(Mv - mn) + sj * __expf(mj - mn);
            Mv = mn;
        }
        lse[(size_t)dir * MDIM + mbase + lane] = Mv + logf(Sv);
    }
}

// ---------------------------------------------------------------------------
// Exact fp32 target logits: one wave per (dir,row).
// fw: tgt = h_fw[b,t] . feats[0, t+1]  (0 if t==T-1)
// bw: tgt = h_bw[b,t] . feats[0, t-1]  (0 if t==0)
// ---------------------------------------------------------------------------
__global__ __launch_bounds__(256) void tgt_kernel(
    const float* __restrict__ feats, const float* __restrict__ hidden,
    float* __restrict__ tgt) {
    int wid  = blockIdx.x * 8 + (threadIdx.x >> 5);
    int lane = threadIdx.x & 31;
    int dir  = wid >> 13;          // / MDIM
    int row  = wid & (MDIM - 1);
    int t    = row & (TT - 1);
    const float* h = hidden + (size_t)row * (2 * DD) + dir * DD;
    const float* x = nullptr;
    if (dir == 0) { if (t < TT - 1) x = feats + (size_t)(t + 1) * DD; }
    else          { if (t > 0)      x = feats + (size_t)(t - 1) * DD; }
    float acc = 0.f;
    if (x) {
#pragma unroll
        for (int k = lane; k < DD; k += 32) acc += h[k] * x[k];
    }
    for (int off = 16; off; off >>= 1) acc += __shfl_xor(acc, off);
    if (lane == 0) tgt[(size_t)dir * MDIM + row] = acc;
}

// ---------------------------------------------------------------------------
// Final losses: out[dir] = sum_rows(lse - tgt) / (seq_len * B)
// ---------------------------------------------------------------------------
__global__ __launch_bounds__(256) void loss_kernel(
    const float* __restrict__ lse, const float* __restrict__ tgt,
    const int* __restrict__ seq_len, float* __restrict__ out) {
    int dir = blockIdx.x;
    __shared__ float red[256];
    float acc = 0.f;
    for (int r = threadIdx.x; r < MDIM; r += 256)
        acc += lse[(size_t)dir * MDIM + r] - tgt[(size_t)dir * MDIM + r];
    red[threadIdx.x] = acc;
    __syncthreads();
    for (int s = 128; s; s >>= 1) {
        if (threadIdx.x < s) red[threadIdx.x] += red[threadIdx.x + s];
        __syncthreads();
    }
    if (threadIdx.x == 0) out[dir] = red[0] / ((float)seq_len[0] * (float)BB);
}

// ---------------------------------------------------------------------------
extern "C" void kernel_launch(void* const* d_in, const int* in_sizes, int n_in,
                              void* d_out, int out_size, void* d_ws, size_t ws_size,
                              hipStream_t stream) {
    const float* feats  = (const float*)d_in[0];
    const float* hidden = (const float*)d_in[1];
    const int*   seqlen = (const int*)d_in[2];
    float*       out    = (float*)d_out;

    char* ws = (char*)d_ws;
    const size_t hbf_bytes = (size_t)2 * 2 * MDIM * DD * sizeof(unsigned short); // 16.78 MB
    const size_t xbf_bytes = (size_t)2 * 2 * NDIM * DD * sizeof(unsigned short); // 16.81 MB
    unsigned short* hbf = (unsigned short*)ws;
    unsigned short* xbf = (unsigned short*)(ws + hbf_bytes);
    float* lse = (float*)(ws + hbf_bytes + xbf_bytes);
    float* tgt = lse + (size_t)2 * MDIM;

    prep_h_kernel<<<(2 * MDIM * DD) / 256, 256, 0, stream>>>(hidden, hbf);
    prep_x_kernel<<<(2 * NDIM * DD) / 256, 256, 0, stream>>>(feats, xbf);
    gemm_lse_kernel<<<dim3(MDIM / 128, 2), 256, 0, stream>>>(hbf, xbf, lse);
    tgt_kernel<<<(2 * MDIM) / 8, 256, 0, stream>>>(feats, hidden, tgt);
    loss_kernel<<<2, 256, 0, stream>>>(lse, tgt, seqlen, out);
}
